// CachedVideoAttention_81870666597074
// MI455X (gfx1250) — matrix operand order repaired and checked by reading, not compiled, not run on hardware
//
#include <hip/hip_runtime.h>
#include <stdint.h>

#define B_   2
#define S_   2048
#define D_   512
#define H_   8
#define DH_  64
#define BH_  (B_*H_)        // 16
#define BHS_ (BH_*S_)       // 32768
#define M_   (B_*S_)        // 4096
#define EPS_ 1e-6f

typedef __attribute__((ext_vector_type(16))) __bf16 v16bf;
typedef __attribute__((ext_vector_type(8)))  __bf16 v8bf;
typedef __attribute__((ext_vector_type(8)))  float  v8f;

__device__ __forceinline__ __bf16 f2bf(float f) {
  union { float f; uint32_t u; } v; v.f = f;
  uint32_t r = v.u + 0x7FFFu + ((v.u >> 16) & 1u);   // round to nearest even
  unsigned short h = (unsigned short)(r >> 16);
  return __builtin_bit_cast(__bf16, h);
}

// A-operand fragment (16xK, 16-bit). Per ISA: lane<16 holds K={0..7,16..23}+k0,
// lane>=16 holds K={8..15,24..31}+k0.  Two 16B loads per lane.
__device__ __forceinline__ v16bf load_fragA(const __bf16* __restrict__ row, int k0, int hf) {
  v8bf lo = *reinterpret_cast<const v8bf*>(row + k0 + hf * 8);
  v8bf hi = *reinterpret_cast<const v8bf*>(row + k0 + 16 + hf * 8);
  v16bf f;
#pragma unroll
  for (int i = 0; i < 8; ++i) { f[i] = lo[i]; f[8 + i] = hi[i]; }
  return f;
}

// B-operand fragment (Kx16), same K order as A: lanes 0-15 hold
// K={0..7,16..23}, lanes 16-31 hold K={8..15,24..31}, from N-major storage
// => two 16B loads.
__device__ __forceinline__ v16bf load_fragB(const __bf16* __restrict__ row, int k0, int hf) {
  v8bf lo = *reinterpret_cast<const v8bf*>(row + k0 + hf * 8);
  v8bf hi = *reinterpret_cast<const v8bf*>(row + k0 + 16 + hf * 8);
  v16bf f;
#pragma unroll
  for (int i = 0; i < 8; ++i) { f[i] = lo[i]; f[8 + i] = hi[i]; }
  return f;
}

__device__ __forceinline__ v8f wmma_bf16(v16bf a, v16bf b, v8f c) {
  return __builtin_amdgcn_wmma_f32_16x16x32_bf16(false, a, false, b, (short)0, c, false, false);
}

// One wave computes a 16x64 f32 tile: A fragment reused across 4 N-tiles.
__device__ __forceinline__ void gemm16x64(const __bf16* __restrict__ A, int lda,
                                          const __bf16* __restrict__ Bt, int ldb,
                                          int m0, int n0, int K,
                                          int hf, int idx, v8f acc[4]) {
  const __bf16* arow = A + (size_t)(m0 + idx) * lda;
  for (int k0 = 0; k0 < K; k0 += 32) {
    v16bf a = load_fragA(arow, k0, hf);
#pragma unroll
    for (int j = 0; j < 4; ++j) {
      const __bf16* brow = Bt + (size_t)(n0 + 16 * j + idx) * ldb;
      v16bf b = load_fragB(brow, k0, hf);
      acc[j] = wmma_bf16(a, b, acc[j]);
    }
  }
}

// ---------------- conversion / transpose kernels ----------------

__global__ void cvt_kernel(const float* __restrict__ in, __bf16* __restrict__ outp, int n) {
  int i = blockIdx.x * 256 + threadIdx.x;
  if (i < n) outp[i] = f2bf(in[i]);
}

// W[K,N] row-major f32  ->  Wt[N,K] (N-major) bf16
__global__ void wtrans_kernel(const float* __restrict__ W, __bf16* __restrict__ Wt, int K, int N) {
  int i = blockIdx.x * 256 + threadIdx.x;
  if (i >= K * N) return;
  int k = i % K, n = i / K;
  Wt[i] = f2bf(W[(size_t)k * N + n]);
}

// v f32 [B,H,S,DH] (seg 2 of qkvf) -> vT bf16 [B,H,DH,S]
__global__ void vtrans_kernel(const float* __restrict__ qkvf, __bf16* __restrict__ vt) {
  int i = blockIdx.x * 256 + threadIdx.x;   // over BHS_*DH_
  int s  = i % S_;
  int t  = i / S_;
  int dh = t % DH_;
  int bh = t / DH_;
  vt[i] = f2bf(qkvf[(size_t)2 * BHS_ * DH_ + ((size_t)bh * S_ + s) * DH_ + dh]);
}

// ---------------- QKV projection GEMM ----------------

__global__ __launch_bounds__(32) void qkv_gemm_kernel(const __bf16* __restrict__ Xb,
                                                      const __bf16* __restrict__ WqT,
                                                      const float* __restrict__ bqkv,
                                                      float* __restrict__ qkvf) {
  const int hf = threadIdx.x >> 4, idx = threadIdx.x & 15;
  const int m0 = blockIdx.x * 16, n0 = blockIdx.y * 64;
  v8f acc[4] = {};
  gemm16x64(Xb, D_, WqT, D_, m0, n0, D_, hf, idx, acc);
  // 64-aligned n0 => this tile covers exactly one (q/k/v segment, head)
  const int seg = n0 / D_;
  const int h   = (n0 % D_) / DH_;
  float* dst = qkvf + (size_t)seg * BHS_ * DH_;
#pragma unroll
  for (int j = 0; j < 4; ++j) {
    const int dh = 16 * j + idx;
    const float bias = bqkv[n0 + 16 * j + idx];
#pragma unroll
    for (int r = 0; r < 8; ++r) {
      const int m = m0 + r + 8 * hf;
      const int b = m / S_, s = m % S_;
      dst[(((size_t)(b * H_ + h)) * S_ + s) * DH_ + dh] = acc[j][r] + bias;
    }
  }
}

// ---------------- RMSNorm (f32 in, bf16 out) ----------------

__global__ __launch_bounds__(256) void rmsnorm_kernel(const float* __restrict__ qkvf,
                                                      const float* __restrict__ wq,
                                                      const float* __restrict__ wk,
                                                      __bf16* __restrict__ qb,
                                                      __bf16* __restrict__ kb) {
  const int wid = threadIdx.x >> 5, lane = threadIdx.x & 31;
  const int row = blockIdx.x * 8 + wid;     // row in [0, B*H*S)
  const int seg = blockIdx.y;               // 0 = q, 1 = k
  const float* w   = seg ? wk : wq;
  const float* src = qkvf + ((size_t)seg * BHS_ + row) * DH_;
  const float x0 = src[2 * lane], x1 = src[2 * lane + 1];
  float ss = x0 * x0 + x1 * x1;
#pragma unroll
  for (int m = 1; m < 32; m <<= 1) ss += __shfl_xor(ss, m, 32);
  const float inv = rsqrtf(ss * (1.0f / DH_) + EPS_);
  __bf16* dst = (seg ? kb : qb) + (size_t)row * DH_;
  dst[2 * lane]     = f2bf(x0 * inv * w[2 * lane]);
  dst[2 * lane + 1] = f2bf(x1 * inv * w[2 * lane + 1]);
}

// ---------------- flash attention: 1 wave per (b*h, 16-query block) ----------------

__global__ __launch_bounds__(32) void attn_kernel(const __bf16* __restrict__ qb,
                                                  const __bf16* __restrict__ kbuf,
                                                  const __bf16* __restrict__ vt,
                                                  __bf16* __restrict__ attn) {
  __shared__ __bf16 pls[16 * 32];
  const int lane = threadIdx.x;
  const int hf = lane >> 4, idx = lane & 15;
  const int bh = blockIdx.y;
  const int q0 = blockIdx.x * 16;
  const __bf16* Q  = qb   + (size_t)bh * S_ * DH_;
  const __bf16* Kp = kbuf + (size_t)bh * S_ * DH_;
  const __bf16* Vt = vt   + (size_t)bh * DH_ * S_;

  const __bf16* qrow = Q + (size_t)(q0 + idx) * DH_;
  v16bf qf0 = load_fragA(qrow, 0, hf);
  v16bf qf1 = load_fragA(qrow, 32, hf);

  v8f o[4] = {};
  float mrow[8], lrow[8];
#pragma unroll
  for (int r = 0; r < 8; ++r) { mrow[r] = -1e30f; lrow[r] = 0.0f; }

  for (int kp0 = 0; kp0 <= q0 + 15; kp0 += 32) {
    // S = Q K^T  (contraction over DH=64 -> two WMMA per 16-key tile)
    v8f s0 = {}, s1 = {};
    {
      const __bf16* k0r = Kp + (size_t)(kp0 + idx) * DH_;
      const __bf16* k1r = Kp + (size_t)(kp0 + 16 + idx) * DH_;
      v16bf kf;
      kf = load_fragB(k0r, 0, hf);  s0 = wmma_bf16(qf0, kf, s0);
      kf = load_fragB(k0r, 32, hf); s0 = wmma_bf16(qf1, kf, s0);
      kf = load_fragB(k1r, 0, hf);  s1 = wmma_bf16(qf0, kf, s1);
      kf = load_fragB(k1r, 32, hf); s1 = wmma_bf16(qf1, kf, s1);
    }
    // causal mask (only needed near the diagonal)
    if (kp0 + 31 > q0) {
      const int kc0 = kp0 + idx, kc1 = kp0 + 16 + idx;
#pragma unroll
      for (int r = 0; r < 8; ++r) {
        const int qr = q0 + r + 8 * hf;
        if (kc0 > qr) s0[r] = -1e30f;
        if (kc1 > qr) s1[r] = -1e30f;
      }
    }
    // online softmax; per-row stats replicated within each 16-lane half
#pragma unroll
    for (int r = 0; r < 8; ++r) {
      float rm = fmaxf(s0[r], s1[r]);
#pragma unroll
      for (int w = 1; w < 16; w <<= 1) rm = fmaxf(rm, __shfl_xor(rm, w, 32));
      const float nm   = fmaxf(mrow[r], rm);
      const float corr = __expf(mrow[r] - nm);
      const float p0 = __expf(s0[r] - nm);
      const float p1 = __expf(s1[r] - nm);
      float ps = p0 + p1;
#pragma unroll
      for (int w = 1; w < 16; w <<= 1) ps += __shfl_xor(ps, w, 32);
      lrow[r] = lrow[r] * corr + ps;
      mrow[r] = nm;
      s0[r] = p0; s1[r] = p1;
#pragma unroll
      for (int j = 0; j < 4; ++j) o[j][r] *= corr;
    }
    // re-layout P [16x32] via LDS: C-layout -> A-fragment layout
#pragma unroll
    for (int r = 0; r < 8; ++r) {
      pls[(r + 8 * hf) * 32 + idx]      = f2bf(s0[r]);
      pls[(r + 8 * hf) * 32 + 16 + idx] = f2bf(s1[r]);
    }
    __syncthreads();
    v16bf pf;
#pragma unroll
    for (int i = 0; i < 8; ++i) {
      pf[i]     = pls[idx * 32 + hf * 8 + i];
      pf[8 + i] = pls[idx * 32 + 16 + hf * 8 + i];
    }
    __syncthreads();
    // O += P V  (contraction over 32 key positions, 4 dh-tiles)
#pragma unroll
    for (int j = 0; j < 4; ++j) {
      v16bf vf = load_fragB(Vt + (size_t)(16 * j + idx) * S_ + kp0, 0, hf);
      o[j] = wmma_bf16(pf, vf, o[j]);
    }
  }

  const int b = bh / H_, h = bh % H_;
#pragma unroll
  for (int j = 0; j < 4; ++j) {
#pragma unroll
    for (int r = 0; r < 8; ++r) {
      const int srow = q0 + r + 8 * hf;
      attn[((size_t)(b * S_ + srow)) * D_ + h * DH_ + 16 * j + idx] = f2bf(o[j][r] / lrow[r]);
    }
  }
}

// ---------------- output projection GEMM ----------------

__global__ __launch_bounds__(32) void oproj_gemm_kernel(const __bf16* __restrict__ Ab,
                                                        const __bf16* __restrict__ WoT,
                                                        const float* __restrict__ bo,
                                                        float* __restrict__ outp) {
  const int hf = threadIdx.x >> 4, idx = threadIdx.x & 15;
  const int m0 = blockIdx.x * 16, n0 = blockIdx.y * 64;
  v8f acc[4] = {};
  gemm16x64(Ab, D_, WoT, D_, m0, n0, D_, hf, idx, acc);
#pragma unroll
  for (int j = 0; j < 4; ++j) {
    const int n = n0 + 16 * j + idx;
    const float bias = bo[n];
#pragma unroll
    for (int r = 0; r < 8; ++r) {
      const int m = m0 + r + 8 * hf;
      outp[(size_t)m * D_ + n] = acc[j][r] + bias;
    }
  }
}

// ---------------- launch ----------------

extern "C" void kernel_launch(void* const* d_in, const int* in_sizes, int n_in,
                              void* d_out, int out_size, void* d_ws, size_t ws_size,
                              hipStream_t stream) {
  const float* x    = (const float*)d_in[0];
  // d_in[1] = causal mask (structure known; unused)
  const float* Wqkv = (const float*)d_in[2];
  const float* bqkv = (const float*)d_in[3];
  const float* Wo   = (const float*)d_in[4];
  const float* bo   = (const float*)d_in[5];
  const float* wq   = (const float*)d_in[6];
  const float* wk   = (const float*)d_in[7];
  float* out = (float*)d_out;

  uint8_t* ws = (uint8_t*)d_ws;
  size_t off = 0;
  auto alloc = [&](size_t bytes) -> void* {
    void* p = ws + off;
    off = (off + bytes + 255) & ~(size_t)255;
    return p;
  };
  __bf16* Xb   = (__bf16*)alloc((size_t)M_ * D_ * 2);            // x in bf16
  __bf16* WqT  = (__bf16*)alloc((size_t)3 * D_ * D_ * 2);        // Wqkv^T bf16 [1536,512]
  __bf16* WoT  = (__bf16*)alloc((size_t)D_ * D_ * 2);            // Wo^T bf16 [512,512]
  float*  qkvf = (float*) alloc((size_t)3 * BHS_ * DH_ * 4);     // qkv f32 [3][B,H,S,DH]
  __bf16* qbuf = (__bf16*)alloc((size_t)BHS_ * DH_ * 2);         // q bf16 [B,H,S,DH]
  __bf16* kbuf = (__bf16*)alloc((size_t)BHS_ * DH_ * 2);         // k bf16 [B,H,S,DH]
  __bf16* vtb  = (__bf16*)alloc((size_t)BHS_ * DH_ * 2);         // v bf16 [B,H,DH,S]
  __bf16* attn = (__bf16*)alloc((size_t)M_ * D_ * 2);            // attn out bf16 [B*S, D]

  cvt_kernel<<<(M_ * D_) / 256, 256, 0, stream>>>(x, Xb, M_ * D_);
  wtrans_kernel<<<(D_ * 3 * D_) / 256, 256, 0, stream>>>(Wqkv, WqT, D_, 3 * D_);
  wtrans_kernel<<<(D_ * D_) / 256, 256, 0, stream>>>(Wo, WoT, D_, D_);

  qkv_gemm_kernel<<<dim3(M_ / 16, (3 * D_) / 64), 32, 0, stream>>>(Xb, WqT, bqkv, qkvf);
  rmsnorm_kernel<<<dim3(BHS_ / 8, 2), 256, 0, stream>>>(qkvf, wq, wk, qbuf, kbuf);
  vtrans_kernel<<<(BHS_ * DH_) / 256, 256, 0, stream>>>(qkvf, vtb);

  attn_kernel<<<dim3(S_ / 16, BH_), 32, 0, stream>>>(qbuf, kbuf, vtb, attn);

  oproj_gemm_kernel<<<dim3(M_ / 16, D_ / 64), 32, 0, stream>>>(attn, WoT, bo, out);
}
